// GlobalCrossAttention_49503793054112
// MI455X (gfx1250) — matrix operand order repaired and checked by reading, not compiled, not run on hardware
//
#include <hip/hip_runtime.h>
#include <hip/hip_bf16.h>

#define BATCH    2
#define NQ       512
#define DIM      256
#define HEADS    8
#define HID      512
#define HEAD_DIM 32
#define DD       8
#define HH       32
#define WW       32
#define NKEY     8192            // DD*HH*WW
#define ATTN_SCALE 0.17677669529663687f  // 32^-0.5

typedef __attribute__((ext_vector_type(16))) __bf16 v16bf;
typedef __attribute__((ext_vector_type(4)))  __bf16 v4bf;
typedef __attribute__((ext_vector_type(8)))  float  v8f;

// K index inside a 16x32 bf16 A-matrix for VGPR-pair i (0..7) and lane half.
__device__ __forceinline__ int a_kidx(int i, int half) {
  return (i < 4) ? (2 * i + 8 * half) : (16 + 2 * (i - 4) + 8 * half);
}

// ---------------------------------------------------------------------------
// fp32 -> bf16 row-major convert (4 elements / thread).
// ---------------------------------------------------------------------------
__global__ void __launch_bounds__(256) cvt_f32_bf16(
    const float* __restrict__ in, __bf16* __restrict__ out)
{
  int i = (blockIdx.x * 256 + threadIdx.x) * 4;
  float4 f = *(const float4*)(in + i);
  v4bf o;
  o[0] = (__bf16)f.x; o[1] = (__bf16)f.y; o[2] = (__bf16)f.z; o[3] = (__bf16)f.w;
  *(v4bf*)(out + i) = o;
}

// W (DIM x DIM fp32 row-major, out = X@W^T) -> Wt bf16 with Wt[k][c] = W[c][k]
// so B-operand loads are contiguous (lane<->k row, 16 consecutive c's).
__global__ void __launch_bounds__(256) cvt_weight_T(
    const float* __restrict__ W, __bf16* __restrict__ Wt)
{
  int i = blockIdx.x * 256 + threadIdx.x;   // 65536 total
  int c = i >> 8, k = i & 255;              // coalesced read over k
  Wt[k * DIM + c] = (__bf16)W[c * DIM + k];
}

// ---------------------------------------------------------------------------
// Y = X @ W^T + b (X bf16 rows x 256, Wt bf16 [k][c]).
// MODE 0: Q  -> bf16 * ATTN_SCALE, [b][h][q][d]
// MODE 1: K  -> bf16 transposed    [b][h][d][n]
// MODE 2: V  -> bf16               [b][h][n][d]
// MODE 3: out-> fp32 row-major     [row][256]
// ---------------------------------------------------------------------------
template <int MODE, int RPI>
__global__ void __launch_bounds__(128) proj_gemm(
    const __bf16* __restrict__ X, const __bf16* __restrict__ Wt,
    const float* __restrict__ bias, void* __restrict__ out)
{
  const int lane = threadIdx.x & 31;
  const int wave = threadIdx.x >> 5;
  const int r0 = blockIdx.x * 16;
  const int c0 = (blockIdx.y * 4 + wave) * 16;
  const int m = lane & 15;
  const int half = lane >> 4;
  const int row = r0 + m;

  const __bf16* xr = X + row * DIM;
  v8f acc = {};
#pragma unroll
  for (int kk = 0; kk < DIM; kk += 32) {
    v16bf a;
#pragma unroll
    for (int i = 0; i < 8; ++i) {
      int k = kk + a_kidx(i, half);
      a[2 * i]     = xr[k];
      a[2 * i + 1] = xr[k + 1];
    }
    v16bf b;
#pragma unroll
    for (int i = 0; i < 8; ++i) {
      int k = kk + a_kidx(i, half);
      b[2 * i]     = Wt[k * DIM + c0 + m];
      b[2 * i + 1] = Wt[(k + 1) * DIM + c0 + m];
    }
    __builtin_prefetch(Wt + (kk + 32 + lane) * DIM + c0, 0, 1);
    acc = __builtin_amdgcn_wmma_f32_16x16x32_bf16(false, a, false, b,
                                                  (short)0, acc, false, false);
  }

  const int n = lane & 15;
  const int c = c0 + n;
  const float bval = bias[c];
  const int h = c >> 5, d = c & 31;
#pragma unroll
  for (int r = 0; r < 8; ++r) {
    int mr = r + 8 * half;
    int grow = r0 + mr;
    float y = acc[r] + bval;
    int img = grow / RPI;        // RPI is a power of two -> shift
    int loc = grow & (RPI - 1);
    if constexpr (MODE == 0) {
      ((__bf16*)out)[((img * HEADS + h) * NQ + loc) * HEAD_DIM + d] =
          (__bf16)(y * ATTN_SCALE);
    } else if constexpr (MODE == 1) {
      ((__bf16*)out)[((img * HEADS + h) * HEAD_DIM + d) * NKEY + loc] = (__bf16)y;
    } else if constexpr (MODE == 2) {
      ((__bf16*)out)[((img * HEADS + h) * NKEY + loc) * HEAD_DIM + d] = (__bf16)y;
    } else {
      ((float*)out)[grow * DIM + c] = y;
    }
  }
}

// ---------------------------------------------------------------------------
// CPB MLP: relu([d0,d1] @ w1^T + b1) @ w2^T, one axis per instantiation.
// ---------------------------------------------------------------------------
template <int NPOS, int AXIS>
__global__ void __launch_bounds__(256) cpb_mlp(
    const float* __restrict__ ref, const float* __restrict__ w1,
    const float* __restrict__ b1, const float* __restrict__ w2,
    float* __restrict__ out, float sc)
{
  int idx = blockIdx.x * 256 + threadIdx.x;
  int h = idx & 7;
  int t = idx >> 3;
  int p = t & (NPOS - 1); t /= NPOS;
  int q = t & (NQ - 1);
  int b = t / NQ;

  const float* rp = ref + (b * NQ + q) * 6;
  float cc = rp[AXIS], ss = rp[AXIS + 3];
  float lo = (cc - 0.5f * ss) * sc;
  float hi = (cc + 0.5f * ss) * sc;
  float pos = ((float)p + 0.5f) * 16.0f;
  float d0 = lo - pos, d1 = hi - pos;

  float acc = 0.f;
  const float* w2h = w2 + h * HID;
  for (int j = 0; j < HID; ++j) {
    float hid = fmaf(d0, w1[2 * j], fmaf(d1, w1[2 * j + 1], b1[j]));
    hid = fmaxf(hid, 0.f);
    acc = fmaf(hid, w2h[j], acc);
  }
  out[((b * NQ + q) * NPOS + p) * HEADS + h] = acc;
}

// ---------------------------------------------------------------------------
// Flash attention: 4 waves / block share one (b,h); each wave owns a 16-query
// tile. K/V chunks (32 keys) are double-buffered in LDS via CDNA5 async
// global->LDS DMA issued by wave 0; separable rpe bias recomposed on the fly.
// Softmax row-sums are accumulated on the matrix pipe (P @ ones WMMA).
// ---------------------------------------------------------------------------
__device__ __forceinline__ void issue_async_kv(
    unsigned long long kbase, unsigned long long vbase, int kc, int lane,
    __bf16 (*ks)[32][32], __bf16 (*vs)[32][32], int buf)
{
  unsigned kvoff = (unsigned)((lane * NKEY + kc) * 2);         // bytes
  unsigned vvoff = (unsigned)(((kc + lane) * HEAD_DIM) * 2);   // bytes
#pragma unroll
  for (int i = 0; i < 4; ++i) {
    unsigned kdst = (unsigned)(unsigned long long)&ks[buf][lane][i * 8];
    asm volatile("global_load_async_to_lds_b128 %0, %1, %2"
                 :: "v"(kdst), "v"(kvoff + i * 16), "s"(kbase) : "memory");
    unsigned vdst = (unsigned)(unsigned long long)&vs[buf][lane][i * 8];
    asm volatile("global_load_async_to_lds_b128 %0, %1, %2"
                 :: "v"(vdst), "v"(vvoff + i * 16), "s"(vbase) : "memory");
  }
}

__global__ void __launch_bounds__(128) flash_attn(
    const __bf16* __restrict__ qbf, const __bf16* __restrict__ kT,
    const __bf16* __restrict__ vbf, const float* __restrict__ rx,
    const float* __restrict__ ry, const float* __restrict__ rz,
    __bf16* __restrict__ outbf)
{
  __shared__ __bf16 ks[2][32][32];     // [buf][d][key]
  __shared__ __bf16 vs[2][32][32];     // [buf][key][d]
  __shared__ __bf16 lds_p[4][16][32];  // per-wave P tile
  __shared__ float  lds_rx[64][32];
  __shared__ float  lds_ry[64][32];
  __shared__ float  lds_rz[64][8];

  const int tid = threadIdx.x;
  const int lane = tid & 31;
  const int wave = tid >> 5;
  int bid = blockIdx.x;
  const int qt = bid & 7; bid >>= 3;   // NQ/64 = 8 block tiles
  const int h = bid & 7;
  const int b = bid >> 3;
  const int q0b = qt * 64;
  const int q0 = q0b + wave * 16;
  const int m = lane & 15, half = lane >> 4, n = lane & 15;

  // Stage separable bias for the block's 64 queries.
  for (int t = tid; t < 64 * 32; t += 128) {
    int ql = t >> 5, x = t & 31;
    lds_rx[ql][x] = rx[((b * NQ + q0b + ql) * WW + x) * HEADS + h];
    lds_ry[ql][x] = ry[((b * NQ + q0b + ql) * HH + x) * HEADS + h];
  }
  for (int t = tid; t < 64 * 8; t += 128) {
    int ql = t >> 3, z = t & 7;
    lds_rz[ql][z] = rz[((b * NQ + q0b + ql) * DD + z) * HEADS + h];
  }

  // Q tile in A layout (pre-scaled bf16).
  v16bf aq;
  {
    const __bf16* qr = qbf + ((b * HEADS + h) * NQ + q0 + m) * HEAD_DIM;
#pragma unroll
    for (int i = 0; i < 8; ++i) {
      int k = a_kidx(i, half);
      aq[2 * i] = qr[k];
      aq[2 * i + 1] = qr[k + 1];
    }
  }

  // Constant all-ones B operand: row-sum of P via the matrix pipe.
  v16bf ones;
#pragma unroll
  for (int i = 0; i < 16; ++i) ones[i] = (__bf16)1.0f;

  const unsigned long long kbase =
      (unsigned long long)(kT + (unsigned long long)(b * HEADS + h) * HEAD_DIM * NKEY);
  const unsigned long long vbase =
      (unsigned long long)(vbf + (unsigned long long)(b * HEADS + h) * NKEY * HEAD_DIM);

  if (wave == 0) {
    issue_async_kv(kbase, vbase, 0, lane, ks, vs, 0);
    issue_async_kv(kbase, vbase, 32, lane, ks, vs, 1);
  }
  __syncthreads();   // rpe staged; async in flight

  v8f o0 = {}, o1 = {}, rs = {};
  float rmax[8];
#pragma unroll
  for (int r = 0; r < 8; ++r) rmax[r] = -1e30f;

  for (int kc = 0; kc < NKEY; kc += 32) {
    const int buf = (kc >> 5) & 1;
    const int y = (kc >> 5) & 31, z = kc >> 10;

    if (wave == 0) {
      if (kc + 32 < NKEY)
        asm volatile("s_wait_asynccnt 0x8" ::: "memory");  // current buf landed
      else
        asm volatile("s_wait_asynccnt 0x0" ::: "memory");
    }
    __syncthreads();   // buffer visible to all waves

    v16bf bk0, bk1, bv0, bv1;
#pragma unroll
    for (int i = 0; i < 8; ++i) {
      int k = a_kidx(i, half);
      bk0[2 * i] = ks[buf][k][n];      bk0[2 * i + 1] = ks[buf][k + 1][n];
      bk1[2 * i] = ks[buf][k][n + 16]; bk1[2 * i + 1] = ks[buf][k + 1][n + 16];
      bv0[2 * i] = vs[buf][k][n];      bv0[2 * i + 1] = vs[buf][k + 1][n];
      bv1[2 * i] = vs[buf][k][n + 16]; bv1[2 * i + 1] = vs[buf][k + 1][n + 16];
    }

    v8f s0 = {}, s1 = {};
    s0 = __builtin_amdgcn_wmma_f32_16x16x32_bf16(false, aq, false, bk0,
                                                 (short)0, s0, false, false);
    s1 = __builtin_amdgcn_wmma_f32_16x16x32_bf16(false, aq, false, bk1,
                                                 (short)0, s1, false, false);

#pragma unroll
    for (int r = 0; r < 8; ++r) {
      int mr = r + 8 * half;
      int qr = wave * 16 + mr;
      float byz = lds_ry[qr][y] + lds_rz[qr][z];
      float s0r = s0[r] + lds_rx[qr][n] + byz;
      float s1r = s1[r] + lds_rx[qr][n + 16] + byz;

      float mx = fmaxf(s0r, s1r);
      mx = fmaxf(mx, __shfl_xor(mx, 1));
      mx = fmaxf(mx, __shfl_xor(mx, 2));
      mx = fmaxf(mx, __shfl_xor(mx, 4));
      mx = fmaxf(mx, __shfl_xor(mx, 8));

      float mnew = fmaxf(rmax[r], mx);
      float alpha = __expf(rmax[r] - mnew);
      float p0 = __expf(s0r - mnew);
      float p1 = __expf(s1r - mnew);
      rmax[r] = mnew;
      o0[r] *= alpha;
      o1[r] *= alpha;
      rs[r] *= alpha;
      lds_p[wave][mr][n]      = (__bf16)p0;
      lds_p[wave][mr][n + 16] = (__bf16)p1;
    }

    // Wave-local LDS visibility is enough for the P round-trip.
    asm volatile("s_wait_dscnt 0x0" ::: "memory");

    v16bf ap;
#pragma unroll
    for (int i = 0; i < 8; ++i) {
      int k = a_kidx(i, half);
      ap[2 * i] = lds_p[wave][m][k];
      ap[2 * i + 1] = lds_p[wave][m][k + 1];
    }

    o0 = __builtin_amdgcn_wmma_f32_16x16x32_bf16(false, ap, false, bv0,
                                                 (short)0, o0, false, false);
    o1 = __builtin_amdgcn_wmma_f32_16x16x32_bf16(false, ap, false, bv1,
                                                 (short)0, o1, false, false);
    rs = __builtin_amdgcn_wmma_f32_16x16x32_bf16(false, ap, false, ones,
                                                 (short)0, rs, false, false);

    __syncthreads();   // all waves done with ks/vs[buf] before refill
    if (wave == 0 && kc + 64 < NKEY)
      issue_async_kv(kbase, vbase, kc + 64, lane, ks, vs, buf);
  }

  // rs columns all hold the row sum; normalize and write bf16 in
  // [b][q][h*32+d] layout (rows of the final projection GEMM).
#pragma unroll
  for (int r = 0; r < 8; ++r) {
    int mr = r + 8 * half;
    float inv = 1.0f / rs[r];
    __bf16* orow = outbf + (b * NQ + q0 + mr) * DIM + h * HEAD_DIM;
    orow[n]      = (__bf16)(o0[r] * inv);
    orow[n + 16] = (__bf16)(o1[r] * inv);
  }
}

// ---------------------------------------------------------------------------
extern "C" void kernel_launch(void* const* d_in, const int* in_sizes, int n_in,
                              void* d_out, int out_size, void* d_ws, size_t ws_size,
                              hipStream_t stream)
{
  (void)in_sizes; (void)n_in; (void)out_size; (void)ws_size;
  const float* query = (const float*)d_in[0];
  const float* refpt = (const float*)d_in[1];
  const float* kin   = (const float*)d_in[2];
  const float* vin   = (const float*)d_in[3];
  const float* wq = (const float*)d_in[4];
  const float* bq = (const float*)d_in[5];
  const float* wk = (const float*)d_in[6];
  const float* bk = (const float*)d_in[7];
  const float* wv = (const float*)d_in[8];
  const float* bv = (const float*)d_in[9];
  const float* wp = (const float*)d_in[10];
  const float* bp = (const float*)d_in[11];
  const float* c1w1 = (const float*)d_in[12];
  const float* c1b1 = (const float*)d_in[13];
  const float* c1w2 = (const float*)d_in[14];
  const float* c2w1 = (const float*)d_in[15];
  const float* c2b1 = (const float*)d_in[16];
  const float* c2w2 = (const float*)d_in[17];
  const float* c3w1 = (const float*)d_in[18];
  const float* c3b1 = (const float*)d_in[19];
  const float* c3w2 = (const float*)d_in[20];

  char* ws = (char*)d_ws;
  __bf16* qbf  = (__bf16*)(ws);                   //  512 KB  q   [b,h,q,d]
  __bf16* kT   = (__bf16*)(ws + 0x0080000);       //    8 MB  kT  [b,h,d,n]
  __bf16* vbf  = (__bf16*)(ws + 0x0880000);       //    8 MB  v   [b,h,n,d]
  float*  rx   = (float*)(ws + 0x1080000);        //    1 MB
  float*  ry   = (float*)(ws + 0x1180000);        //    1 MB
  float*  rz   = (float*)(ws + 0x1280000);        //  256 KB
  __bf16* aobf = (__bf16*)(ws + 0x12C0000);       //  512 KB  attention out
  __bf16* xqbf = (__bf16*)(ws + 0x1340000);       //  512 KB  query bf16
  __bf16* xkbf = (__bf16*)(ws + 0x13C0000);       //    4 MB  k input bf16
  __bf16* xvbf = (__bf16*)(ws + 0x17C0000);       //    4 MB  v input bf16
  __bf16* wqT  = (__bf16*)(ws + 0x1BC0000);       //  128 KB each
  __bf16* wkT  = (__bf16*)(ws + 0x1BE0000);
  __bf16* wvT  = (__bf16*)(ws + 0x1C00000);
  __bf16* wpT  = (__bf16*)(ws + 0x1C20000);

  // Stage inputs/weights as bf16 (weights transposed into WMMA B layout).
  cvt_f32_bf16<<<BATCH * NQ * DIM / 1024, 256, 0, stream>>>(query, xqbf);
  cvt_f32_bf16<<<BATCH * NKEY * DIM / 1024, 256, 0, stream>>>(kin, xkbf);
  cvt_f32_bf16<<<BATCH * NKEY * DIM / 1024, 256, 0, stream>>>(vin, xvbf);
  cvt_weight_T<<<DIM * DIM / 256, 256, 0, stream>>>(wq, wqT);
  cvt_weight_T<<<DIM * DIM / 256, 256, 0, stream>>>(wk, wkT);
  cvt_weight_T<<<DIM * DIM / 256, 256, 0, stream>>>(wv, wvT);
  cvt_weight_T<<<DIM * DIM / 256, 256, 0, stream>>>(wp, wpT);

  const dim3 blk(128);
  proj_gemm<0, NQ><<<dim3(BATCH * NQ / 16, 4), blk, 0, stream>>>(xqbf, wqT, bq, qbf);
  proj_gemm<1, NKEY><<<dim3(BATCH * NKEY / 16, 4), blk, 0, stream>>>(xkbf, wkT, bk, kT);
  proj_gemm<2, NKEY><<<dim3(BATCH * NKEY / 16, 4), blk, 0, stream>>>(xvbf, wvT, bv, vbf);

  cpb_mlp<WW, 0><<<(BATCH * NQ * WW * HEADS) / 256, 256, 0, stream>>>(refpt, c1w1, c1b1, c1w2, rx, (float)(WW * 16));
  cpb_mlp<HH, 1><<<(BATCH * NQ * HH * HEADS) / 256, 256, 0, stream>>>(refpt, c2w1, c2b1, c2w2, ry, (float)(HH * 16));
  cpb_mlp<DD, 2><<<(BATCH * NQ * DD * HEADS) / 256, 256, 0, stream>>>(refpt, c3w1, c3b1, c3w2, rz, (float)(DD * 16));

  flash_attn<<<BATCH * HEADS * (NQ / 64), 128, 0, stream>>>(qbf, kT, vbf, rx, ry, rz, aobf);

  proj_gemm<3, NQ><<<dim3(BATCH * NQ / 16, 4), blk, 0, stream>>>(aobf, wpT, bp, (float*)d_out);
}
